// DecoderLayer_76046690943493
// MI455X (gfx1250) — matrix operand order repaired and checked by reading, not compiled, not run on hardware
//
#include <hip/hip_runtime.h>

// ---------------------------------------------------------------------------
// MI455X (gfx1250) decoder layer, v3.
// - All GEMM operands pre-converted to bf16 once (weights transposed to [N,K]
//   so every B operand is N-major => LDS fills are pure copies).
// - GEMM inner loop: global_load_async_to_lds_b128 (CDNA5 async DMA) double
//   buffered, 8x v_wmma_f32_16x16x32_bf16 per 32-K chunk, zero VALU in fill.
// - Epilogue writes f32 and/or bf16 (and bf16-transposed for V) so consumers
//   always read bf16.
// ---------------------------------------------------------------------------

typedef float          v8f  __attribute__((ext_vector_type(8)));
typedef unsigned short v8us __attribute__((ext_vector_type(8)));
typedef __bf16         v16bf __attribute__((ext_vector_type(16)));

struct BfPack { v8us lo, hi; };

static __device__ __forceinline__ v16bf make_frag(v8us lo, v8us hi) {
  BfPack p{lo, hi};
  return __builtin_bit_cast(v16bf, p);
}

// f32 -> bf16, round-to-nearest (ties away): 2 VALU
static __device__ __forceinline__ unsigned short f2bf(float f) {
  return (unsigned short)((__float_as_uint(f) + 0x8000u) >> 16);
}
// two f32 -> packed bf16x2 dword: 2 adds + one v_perm_b32
static __device__ __forceinline__ unsigned pack_bf16(float a, float b) {
  unsigned ua = __float_as_uint(a) + 0x8000u;
  unsigned ub = __float_as_uint(b) + 0x8000u;
  return __builtin_amdgcn_perm(ub, ua, 0x07060302u);
}

// ---- CDNA5 async copy: 16B global -> LDS, tracked by ASYNCcnt ----
static __device__ __forceinline__ void async_copy16(unsigned lds_byte_off,
                                                    const void* sbase,
                                                    unsigned gbyte_off) {
  asm volatile("global_load_async_to_lds_b128 %0, %1, %2"
               :: "v"(lds_byte_off), "v"(gbyte_off), "s"(sbase) : "memory");
}
static __device__ __forceinline__ void wait_async0() {
  asm volatile("s_wait_asynccnt 0x0" ::: "memory");
}
static __device__ __forceinline__ unsigned lds_addr32(const void* p) {
  return (unsigned)(unsigned long long)p;   // generic LDS ptr: low 32 bits = LDS offset
}

#define GF_RELU   2
#define GF_CAUSAL 4

#define BM 128
#define BN 64
#define LDS_S 40          // LDS row stride (elems): 16B-aligned frag loads, bank spread
#define SA_BUF (BM * LDS_S)
#define SB_BUF (BN * LDS_S)

// A: bf16 [M,K] row-major. B: bf16 [N,K] row-major (i.e. always B^T layout).
// Outputs (each nullable): Cf  f32 [M,N];  Cb bf16 [M,N];
//                          Cbt bf16 "per-1024-row transposed": [m>>10][n][m&1023]
__global__ __launch_bounds__(128)
void gemm_bf16_async(const unsigned short* __restrict__ A, long long lda, long long sAb, long long sAh,
                     const unsigned short* __restrict__ B, long long ldb, long long sBb, long long sBh,
                     const float* __restrict__ bias,
                     float* __restrict__ Cf, unsigned short* __restrict__ Cb,
                     unsigned short* __restrict__ Cbt,
                     long long ldc, long long sCb, long long sCh,
                     int K, int nH, float scale, int flags)
{
  __shared__ __align__(16) unsigned short sA[2 * SA_BUF];
  __shared__ __align__(16) unsigned short sB[2 * SB_BUF];

  const int z  = blockIdx.z;
  const int zb = z / nH, zh = z % nH;
  A += (long long)zb * sAb + (long long)zh * sAh;
  B += (long long)zb * sBb + (long long)zh * sBh;
  const long long coff = (long long)zb * sCb + (long long)zh * sCh;

  const int n0 = blockIdx.x * BN;
  const int m0 = blockIdx.y * BM;
  const int tid = threadIdx.x;
  const int lane = tid & 31, wid = tid >> 5;
  const int rbase = (wid >> 1) * 64;   // wave row offset in 128x64 tile
  const int cbase = (wid & 1) * 32;    // wave col offset
  const int lm = lane & 15, lh = lane >> 4;

  const unsigned sA0 = lds_addr32(&sA[0]);
  const unsigned sB0 = lds_addr32(&sB[0]);
  const unsigned short* Abase = A + (long long)m0 * lda;
  const unsigned short* Bbase = B + (long long)n0 * ldb;

  // issue the async fills for one 32-wide K chunk into LDS buffer `buf`
  auto issue_chunk = [&](int k0, int buf) {
    const unsigned short* Ak = Abase + k0;
    const unsigned short* Bk = Bbase + k0;
    const unsigned oa = (unsigned)(buf * SA_BUF) * 2 + sA0;
    const unsigned ob = (unsigned)(buf * SB_BUF) * 2 + sB0;
#pragma unroll
    for (int it = 0; it < 4; ++it) {               // A: 128 rows x 4 x 16B
      int slot = tid + it * 128;
      int m = slot >> 2, cc = slot & 3;
      async_copy16(oa + (unsigned)(m * LDS_S + cc * 8) * 2, Ak,
                   (unsigned)((m * (int)lda + cc * 8) * 2));
    }
#pragma unroll
    for (int it = 0; it < 2; ++it) {               // B: 64 rows x 4 x 16B
      int slot = tid + it * 128;
      int n = slot >> 2, cc = slot & 3;
      async_copy16(ob + (unsigned)(n * LDS_S + cc * 8) * 2, Bk,
                   (unsigned)((n * (int)ldb + cc * 8) * 2));
    }
  };

  v8f acc[4][2] = {};
  const int nch = K >> 5;

  issue_chunk(0, 0);
  wait_async0();
  __syncthreads();

  for (int c = 0; c < nch; ++c) {
    const int cur = c & 1;
    if (c + 1 < nch) issue_chunk((c + 1) << 5, 1 - cur);  // DMA next chunk behind WMMAs

    // A 16x32 frag: lane lm = row; lh picks k-half (VGPR0-3 k=lh*8+0..7, VGPR4-7 +16)
    // B 32x16 frag: lane lm = col; lh picks k-half (VGPR0-3 k=lh*8+0..7, VGPR4-7 +16)
    const int oa = cur * SA_BUF, ob = cur * SB_BUF;
    v16bf af[4], bfr[2];
#pragma unroll
    for (int mi = 0; mi < 4; ++mi) {
      int r = oa + (rbase + mi * 16 + lm) * LDS_S + lh * 8;
      af[mi] = make_frag(*(const v8us*)&sA[r], *(const v8us*)&sA[r + 16]);
    }
#pragma unroll
    for (int nj = 0; nj < 2; ++nj) {
      int r = ob + (cbase + nj * 16 + lm) * LDS_S + lh * 8;
      bfr[nj] = make_frag(*(const v8us*)&sB[r], *(const v8us*)&sB[r + 16]);
    }
#pragma unroll
    for (int mi = 0; mi < 4; ++mi)
#pragma unroll
      for (int nj = 0; nj < 2; ++nj)
        acc[mi][nj] = __builtin_amdgcn_wmma_f32_16x16x32_bf16(
            false, af[mi], false, bfr[nj], (short)0, acc[mi][nj], false, false);

    wait_async0();
    __syncthreads();
  }

  // epilogue: VGPR r -> row r (lanes 0-15) / r+8 (lanes 16-31); lane%16 = col
#pragma unroll
  for (int mi = 0; mi < 4; ++mi)
#pragma unroll
    for (int nj = 0; nj < 2; ++nj)
#pragma unroll
      for (int r = 0; r < 8; ++r) {
        int m = m0 + rbase + mi * 16 + lh * 8 + r;
        int n = n0 + cbase + nj * 16 + lm;
        float v = acc[mi][nj][r] * scale;
        if (bias) v += bias[n];
        if (flags & GF_RELU) v = fmaxf(v, 0.0f);
        if ((flags & GF_CAUSAL) && n > m) v += -10000.0f;   // look_ahead_mask * -1e4
        long long ci = coff + (long long)m * ldc + n;
        if (Cf)  Cf[ci] = v;
        if (Cb)  Cb[ci] = f2bf(v);
        if (Cbt) Cbt[coff + (((long long)(m >> 10) * ldc + n) << 10) + (m & 1023)] = f2bf(v);
      }
}

// elementwise f32 -> packed bf16 pairs
__global__ __launch_bounds__(256)
void cvt_bf16_pairs(const float* __restrict__ in, unsigned* __restrict__ out)
{
  long long i = (long long)blockIdx.x * 256 + threadIdx.x;
  float2 f = ((const float2*)in)[i];
  out[i] = pack_bf16(f.x, f.y);
}

// W [K,N] f32 -> Wt [N,K] bf16 (32x32 LDS tile transpose)
__global__ __launch_bounds__(256)
void transpose_cvt_bf16(const float* __restrict__ in, unsigned short* __restrict__ out,
                        int Kd, int Nd)
{
  __shared__ float t[32][33];
  const int tx = threadIdx.x, ty = threadIdx.y;       // 32 x 8
  const int gn = blockIdx.x * 32, gk = blockIdx.y * 32;
#pragma unroll
  for (int j = 0; j < 4; ++j)
    t[ty + j * 8][tx] = in[(long long)(gk + ty + j * 8) * Nd + (gn + tx)];
  __syncthreads();
#pragma unroll
  for (int j = 0; j < 4; ++j)
    out[(long long)(gn + ty + j * 8) * Kd + (gk + tx)] = f2bf(t[tx][ty + j * 8]);
}

// In-place f32 row softmax (rows of 1024) + bf16 copy for the PV GEMM.
__global__ __launch_bounds__(256)
void softmax1024_dual(float* __restrict__ data, unsigned short* __restrict__ outb)
{
  const int wid = threadIdx.x >> 5, lane = threadIdx.x & 31;
  const size_t row = (size_t)blockIdx.x * 8 + wid;
  float* p = data + row * 1024;
  unsigned short* pb = outb + row * 1024;
  float v[32];
  float mx = -3.4e38f;
#pragma unroll
  for (int i = 0; i < 32; ++i) { v[i] = p[lane + i * 32]; mx = fmaxf(mx, v[i]); }
#pragma unroll
  for (int off = 16; off > 0; off >>= 1) mx = fmaxf(mx, __shfl_xor(mx, off, 32));
  float s = 0.0f;
#pragma unroll
  for (int i = 0; i < 32; ++i) { v[i] = __expf(v[i] - mx); s += v[i]; }
#pragma unroll
  for (int off = 16; off > 0; off >>= 1) s += __shfl_xor(s, off, 32);
  float inv = 1.0f / s;
#pragma unroll
  for (int i = 0; i < 32; ++i) {
    float r = v[i] * inv;
    p[lane + i * 32] = r;
    pb[lane + i * 32] = f2bf(r);
  }
}

// out = LayerNorm(x + y); f32 out + optional bf16 copy.
__global__ __launch_bounds__(256)
void add_layernorm512_dual(const float* __restrict__ x, const float* __restrict__ y,
                           const float* __restrict__ gamma, const float* __restrict__ beta,
                           float* __restrict__ out, unsigned short* __restrict__ outb)
{
  const int wid = threadIdx.x >> 5, lane = threadIdx.x & 31;
  const size_t row = (size_t)blockIdx.x * 8 + wid;
  const float* px = x + row * 512;
  const float* py = y + row * 512;
  float v[16];
  float s = 0.0f;
#pragma unroll
  for (int i = 0; i < 16; ++i) { v[i] = px[lane + i * 32] + py[lane + i * 32]; s += v[i]; }
#pragma unroll
  for (int off = 16; off > 0; off >>= 1) s += __shfl_xor(s, off, 32);
  float mu = s * (1.0f / 512.0f);
  float q = 0.0f;
#pragma unroll
  for (int i = 0; i < 16; ++i) { float d = v[i] - mu; q += d * d; }
#pragma unroll
  for (int off = 16; off > 0; off >>= 1) q += __shfl_xor(q, off, 32);
  float rstd = rsqrtf(q * (1.0f / 512.0f) + 1e-5f);
#pragma unroll
  for (int i = 0; i < 16; ++i) {
    int c = lane + i * 32;
    float r = (v[i] - mu) * rstd * gamma[c] + beta[c];
    out[row * 512 + c] = r;
    if (outb) outb[row * 512 + c] = f2bf(r);
  }
}

// ---------------------------------------------------------------------------

static void gemm(hipStream_t st,
                 const unsigned short* A, long long lda, long long sAb, long long sAh,
                 const unsigned short* B, long long ldb, long long sBb, long long sBh,
                 const float* bias, float* Cf, unsigned short* Cb, unsigned short* Cbt,
                 long long ldc, long long sCb, long long sCh,
                 int M, int N, int K, int nH, int batches, float scale, int flags)
{
  dim3 g((unsigned)(N / BN), (unsigned)(M / BM), (unsigned)batches), b(128, 1, 1);
  gemm_bf16_async<<<g, b, 0, st>>>(A, lda, sAb, sAh, B, ldb, sBb, sBh, bias,
                                   Cf, Cb, Cbt, ldc, sCb, sCh, K, nH, scale, flags);
}

extern "C" void kernel_launch(void* const* d_in, const int* in_sizes, int n_in,
                              void* d_out, int out_size, void* d_ws, size_t ws_size,
                              hipStream_t stream)
{
  (void)in_sizes; (void)n_in; (void)out_size; (void)ws_size;

  const long long D = 512, T = 1024, Bn = 8, H = 8, DH = 64, F = 2048;
  const long long BT = Bn * T;                 // 8192
  const long long BD = BT * D;                 // 4,194,304
  const long long ATT = Bn * H * T * T;        // 67,108,864

  const float* x    = (const float*)d_in[0];
  const float* enc  = (const float*)d_in[1];
  const float* wq1 = (const float*)d_in[4];  const float* bq1 = (const float*)d_in[5];
  const float* wk1 = (const float*)d_in[6];  const float* bk1 = (const float*)d_in[7];
  const float* wv1 = (const float*)d_in[8];  const float* bv1 = (const float*)d_in[9];
  const float* wo1 = (const float*)d_in[10]; const float* bo1 = (const float*)d_in[11];
  const float* wq2 = (const float*)d_in[12]; const float* bq2 = (const float*)d_in[13];
  const float* wk2 = (const float*)d_in[14]; const float* bk2 = (const float*)d_in[15];
  const float* wv2 = (const float*)d_in[16]; const float* bv2 = (const float*)d_in[17];
  const float* wo2 = (const float*)d_in[18]; const float* bo2 = (const float*)d_in[19];
  const float* wf1 = (const float*)d_in[20]; const float* bf1 = (const float*)d_in[21];
  const float* wf2 = (const float*)d_in[22]; const float* bf2 = (const float*)d_in[23];
  const float* g1  = (const float*)d_in[24]; const float* be1 = (const float*)d_in[25];
  const float* g2  = (const float*)d_in[26]; const float* be2 = (const float*)d_in[27];
  const float* g3  = (const float*)d_in[28]; const float* be3 = (const float*)d_in[29];

  float* out3 = (float*)d_out;
  float* saw  = out3 + BD;
  float* caw  = saw + ATT;

  // ---- workspace: f32 first, then bf16 (ushort) ----
  float* resf = (float*)d_ws;          // [BT,D]
  float* o1f  = resf + BD;
  float* o2f  = o1f + BD;
  unsigned short* us = (unsigned short*)(o2f + BD);
  unsigned short* xbf  = us;           us += BD;      // x bf16
  unsigned short* encbf= us;           us += BD;
  unsigned short* qbf  = us;           us += BD;
  unsigned short* kbf  = us;           us += BD;
  unsigned short* vtbf = us;           us += BD;      // [b][hd][t]
  unsigned short* tbf  = us;           us += BD;      // attn output
  unsigned short* o1bf = us;           us += BD;
  unsigned short* o2bf = us;           us += BD;
  unsigned short* ffbf = us;           us += BT * F;  // [BT,F]
  unsigned short* pbf  = us;           us += ATT;     // bf16 probs
  unsigned short* wt   = us;                          // transposed bf16 weights
  unsigned short* wq1t = wt;                wt += D * D;
  unsigned short* wk1t = wt;                wt += D * D;
  unsigned short* wv1t = wt;                wt += D * D;
  unsigned short* wo1t = wt;                wt += D * D;
  unsigned short* wq2t = wt;                wt += D * D;
  unsigned short* wk2t = wt;                wt += D * D;
  unsigned short* wv2t = wt;                wt += D * D;
  unsigned short* wo2t = wt;                wt += D * D;
  unsigned short* wf1t = wt;                wt += F * D;   // [F,D]
  unsigned short* wf2t = wt;                                // [D,F]

  const float iscale = 0.125f;               // 1/sqrt(64)
  const long long sTD = T * D;
  const long long sAT = (long long)T * T;
  const dim3 tb32(32, 8, 1);

  // ---- one-time operand conversion ----
  cvt_bf16_pairs<<<dim3((unsigned)(BD / 512)), dim3(256), 0, stream>>>(x,   (unsigned*)xbf);
  cvt_bf16_pairs<<<dim3((unsigned)(BD / 512)), dim3(256), 0, stream>>>(enc, (unsigned*)encbf);
  transpose_cvt_bf16<<<dim3(16, 16), tb32, 0, stream>>>(wq1, wq1t, (int)D, (int)D);
  transpose_cvt_bf16<<<dim3(16, 16), tb32, 0, stream>>>(wk1, wk1t, (int)D, (int)D);
  transpose_cvt_bf16<<<dim3(16, 16), tb32, 0, stream>>>(wv1, wv1t, (int)D, (int)D);
  transpose_cvt_bf16<<<dim3(16, 16), tb32, 0, stream>>>(wo1, wo1t, (int)D, (int)D);
  transpose_cvt_bf16<<<dim3(16, 16), tb32, 0, stream>>>(wq2, wq2t, (int)D, (int)D);
  transpose_cvt_bf16<<<dim3(16, 16), tb32, 0, stream>>>(wk2, wk2t, (int)D, (int)D);
  transpose_cvt_bf16<<<dim3(16, 16), tb32, 0, stream>>>(wv2, wv2t, (int)D, (int)D);
  transpose_cvt_bf16<<<dim3(16, 16), tb32, 0, stream>>>(wo2, wo2t, (int)D, (int)D);
  transpose_cvt_bf16<<<dim3(64, 16), tb32, 0, stream>>>(wf1, wf1t, (int)D, (int)F);  // -> [F,D]
  transpose_cvt_bf16<<<dim3(16, 64), tb32, 0, stream>>>(wf2, wf2t, (int)F, (int)D);  // -> [D,F]

  // ---------------- self-attention ----------------
  gemm(stream, xbf, D, 0, 0, wq1t, D, 0, 0, bq1, nullptr, qbf, nullptr, D, 0, 0,
       (int)BT, (int)D, (int)D, 1, 1, 1.0f, 0);
  gemm(stream, xbf, D, 0, 0, wk1t, D, 0, 0, bk1, nullptr, kbf, nullptr, D, 0, 0,
       (int)BT, (int)D, (int)D, 1, 1, 1.0f, 0);
  gemm(stream, xbf, D, 0, 0, wv1t, D, 0, 0, bv1, nullptr, nullptr, vtbf, D, 0, 0,
       (int)BT, (int)D, (int)D, 1, 1, 1.0f, 0);
  gemm(stream, qbf, D, sTD, DH, kbf, D, sTD, DH, nullptr,
       saw, nullptr, nullptr, T, (long long)H * sAT, sAT,
       (int)T, (int)T, (int)DH, (int)H, (int)(Bn * H), iscale, GF_CAUSAL);
  softmax1024_dual<<<dim3((unsigned)(Bn * H * T / 8)), dim3(256), 0, stream>>>(saw, pbf);
  gemm(stream, pbf, T, (long long)H * sAT, sAT, vtbf, T, (long long)D * T, DH * T, nullptr,
       nullptr, tbf, nullptr, D, sTD, DH,
       (int)T, (int)DH, (int)T, (int)H, (int)(Bn * H), 1.0f, 0);
  gemm(stream, tbf, D, 0, 0, wo1t, D, 0, 0, bo1, resf, nullptr, nullptr, D, 0, 0,
       (int)BT, (int)D, (int)D, 1, 1, 1.0f, 0);
  add_layernorm512_dual<<<dim3((unsigned)(BT / 8)), dim3(256), 0, stream>>>(x, resf, g1, be1, o1f, o1bf);

  // ---------------- cross-attention ----------------
  gemm(stream, o1bf, D, 0, 0, wq2t, D, 0, 0, bq2, nullptr, qbf, nullptr, D, 0, 0,
       (int)BT, (int)D, (int)D, 1, 1, 1.0f, 0);
  gemm(stream, encbf, D, 0, 0, wk2t, D, 0, 0, bk2, nullptr, kbf, nullptr, D, 0, 0,
       (int)BT, (int)D, (int)D, 1, 1, 1.0f, 0);
  gemm(stream, encbf, D, 0, 0, wv2t, D, 0, 0, bv2, nullptr, nullptr, vtbf, D, 0, 0,
       (int)BT, (int)D, (int)D, 1, 1, 1.0f, 0);
  gemm(stream, qbf, D, sTD, DH, kbf, D, sTD, DH, nullptr,
       caw, nullptr, nullptr, T, (long long)H * sAT, sAT,
       (int)T, (int)T, (int)DH, (int)H, (int)(Bn * H), iscale, 0);   // padding mask == 0
  softmax1024_dual<<<dim3((unsigned)(Bn * H * T / 8)), dim3(256), 0, stream>>>(caw, pbf);
  gemm(stream, pbf, T, (long long)H * sAT, sAT, vtbf, T, (long long)D * T, DH * T, nullptr,
       nullptr, tbf, nullptr, D, sTD, DH,
       (int)T, (int)DH, (int)T, (int)H, (int)(Bn * H), 1.0f, 0);
  gemm(stream, tbf, D, 0, 0, wo2t, D, 0, 0, bo2, resf, nullptr, nullptr, D, 0, 0,
       (int)BT, (int)D, (int)D, 1, 1, 1.0f, 0);
  add_layernorm512_dual<<<dim3((unsigned)(BT / 8)), dim3(256), 0, stream>>>(o1f, resf, g2, be2, o2f, o2bf);

  // ---------------- FFN ----------------
  gemm(stream, o2bf, D, 0, 0, wf1t, D, 0, 0, bf1, nullptr, ffbf, nullptr, F, 0, 0,
       (int)BT, (int)F, (int)D, 1, 1, 1.0f, GF_RELU);
  gemm(stream, ffbf, F, 0, 0, wf2t, F, 0, 0, bf2, resf, nullptr, nullptr, D, 0, 0,
       (int)BT, (int)D, (int)F, 1, 1, 1.0f, 0);
  add_layernorm512_dual<<<dim3((unsigned)(BT / 8)), dim3(256), 0, stream>>>(o2f, resf, g3, be3, out3, nullptr);
}
